// VarianceAdaptor_86517821216336
// MI455X (gfx1250) — matrix operand order repaired and checked by reading, not compiled, not run on hardware
//
#include <hip/hip_runtime.h>
#include <hip/hip_bf16.h>

// Problem constants (from reference)
#define BB  32
#define SS  512
#define DD  256
#define TT  2048
#define FSZ 256
#define EPS 1e-5f

typedef __bf16 bf16;
typedef __attribute__((ext_vector_type(16))) __bf16 v16bf;
typedef __attribute__((ext_vector_type(8)))  __bf16 v8bf;
typedef __attribute__((ext_vector_type(4)))  __bf16 v4bf;
typedef __attribute__((ext_vector_type(8)))  float  v8f;

// ---------------------------------------------------------------------------
// Fused conv1d(K=3, Cin=256 -> 256) + bias + ReLU + LayerNorm(256)
// Optional fused final linear (256 -> 1) + mask for predictor output.
// One block = 8 waves, computes 32 timesteps x 256 channels.
//   wave grid: 2 (M) x 4 (N); each wave owns 4 C fragments (16x16) along N.
// ---------------------------------------------------------------------------
union __attribute__((aligned(32))) SMem {
    struct {
        bf16 A[34][32];          // activation strip: rows t0-1 .. t0+32, one 32-ch chunk
        bf16 Bw[3][256][32];     // weights for 3 taps: [k][f][d-chunk]
    } in;                        // 2176 + 49152 bytes
    float S[32][256];            // LN staging (32768 bytes)
};

__global__ __launch_bounds__(256)
void conv_ln_kernel(const bf16* __restrict__ xin, int Tlen,
                    const bf16* __restrict__ wpack,      // [k][f][d] bf16
                    const float* __restrict__ bias,
                    const float* __restrict__ gamma,
                    const float* __restrict__ beta,
                    bf16* __restrict__ hout,             // may be null
                    float* __restrict__ pred,            // may be null
                    const float* __restrict__ lw,
                    const float* __restrict__ lb,
                    const unsigned char* __restrict__ mask)
{
    __shared__ SMem sm;

    const int bi   = blockIdx.y;
    const int t0   = blockIdx.x * 32;
    const int tid  = threadIdx.x;
    const int lane = tid & 31;
    const int wid  = tid >> 5;
    const int wm   = wid >> 2;      // 0..1 : which 16-row block
    const int wn   = wid & 3;       // 0..3 : which 64-channel block
    const int nrow = lane & 15;     // M (for A/C) or N (for B/C)
    const int half = lane >> 4;     // lane half select

    v8f acc[4] = {};

    for (int dc = 0; dc < DD / 32; ++dc) {
        // ---- stage activation strip (34 rows x 32 bf16), zero-padded ----
        if (tid < 136) {
            int row = tid >> 2;
            int ck  = tid & 3;
            int g   = t0 - 1 + row;
            v8bf val = {};
            if (g >= 0 && g < Tlen)
                val = *(const v8bf*)(xin + ((size_t)bi * Tlen + g) * DD + dc * 32 + ck * 8);
            *(v8bf*)(&sm.in.A[row][ck * 8]) = val;
        }
        // ---- stage weights for 3 taps (3 x 256 x 32 bf16) ----
        #pragma unroll
        for (int it = 0; it < 12; ++it) {
            int tsk = tid + it * 256;              // 0..3071
            int k = tsk >> 10;
            int f = (tsk >> 2) & 255;
            int c = tsk & 3;
            v8bf w = *(const v8bf*)(wpack + ((size_t)(k * FSZ + f)) * DD + dc * 32 + c * 8);
            *(v8bf*)(&sm.in.Bw[k][f][c * 8]) = w;
        }
        __syncthreads();

        // ---- WMMA: 3 taps x 4 N-fragments ----
        #pragma unroll
        for (int k = 0; k < 3; ++k) {
            union { v16bf v; v8bf h[2]; } a;
            int arow = wm * 16 + nrow + k;         // output row m uses input rows m..m+2
            a.h[0] = *(const v8bf*)(&sm.in.A[arow][8 * half]);
            a.h[1] = *(const v8bf*)(&sm.in.A[arow][16 + 8 * half]);
            #pragma unroll
            for (int fi = 0; fi < 4; ++fi) {
                int f = wn * 64 + fi * 16 + nrow;  // N = lane%16
                union { v16bf v; v8bf h[2]; } bfrag;
                bfrag.h[0] = *(const v8bf*)(&sm.in.Bw[k][f][8 * half]);
                bfrag.h[1] = *(const v8bf*)(&sm.in.Bw[k][f][16 + 8 * half]);
                acc[fi] = __builtin_amdgcn_wmma_f32_16x16x32_bf16(
                    false, a.v, false, bfrag.v, (short)0, acc[fi], false, false);
            }
        }
        __syncthreads();
    }

    // ---- bias + ReLU into LDS staging (layout: [row][channel]) ----
    #pragma unroll
    for (int fi = 0; fi < 4; ++fi) {
        int ch = wn * 64 + fi * 16 + nrow;
        float bs = bias[ch];
        #pragma unroll
        for (int j = 0; j < 8; ++j) {
            int r = wm * 16 + 8 * half + j;        // C frag: M = j + 8*half
            float v = acc[fi][j] + bs;
            sm.S[r][ch] = v > 0.f ? v : 0.f;
        }
    }
    __syncthreads();

    // ---- LayerNorm over 256 channels: 8 threads per row ----
    const int r = tid >> 3;
    const int p = tid & 7;
    float s = 0.f, q = 0.f;
    float vals[32];
    #pragma unroll
    for (int i = 0; i < 32; ++i) {
        float v = sm.S[r][p * 32 + i];
        vals[i] = v; s += v; q += v * v;
    }
    #pragma unroll
    for (int m = 1; m < 8; m <<= 1) {
        s += __shfl_xor(s, m, 32);
        q += __shfl_xor(q, m, 32);
    }
    float mean = s * (1.f / 256.f);
    float var  = q * (1.f / 256.f) - mean * mean;
    float rstd = rsqrtf(var + EPS);
    int t = t0 + r;

    if (hout) {
        // straight-line path: normalize + pack into 4 x 16B vector stores
        union { v8bf v[4]; bf16 e[32]; } yb;
        #pragma unroll
        for (int i = 0; i < 32; ++i) {
            int ch = p * 32 + i;
            float y = (vals[i] - mean) * rstd * gamma[ch] + beta[ch];
            yb.e[i] = (bf16)y;
        }
        v8bf* dst = (v8bf*)(hout + ((size_t)bi * Tlen + t) * FSZ + p * 32);
        #pragma unroll
        for (int c = 0; c < 4; ++c) dst[c] = yb.v[c];
    } else {
        // predictor path: fused linear (256 -> 1) + mask
        float dacc = 0.f;
        #pragma unroll
        for (int i = 0; i < 32; ++i) {
            int ch = p * 32 + i;
            float y = (vals[i] - mean) * rstd * gamma[ch] + beta[ch];
            dacc += y * lw[ch];
        }
        #pragma unroll
        for (int m = 1; m < 8; m <<= 1) dacc += __shfl_xor(dacc, m, 32);
        if (p == 0) {
            float o = dacc + lb[0];
            if (mask[(size_t)bi * Tlen + t]) o = 0.f;
            pred[(size_t)bi * Tlen + t] = o;
        }
    }
}

// ---------------------------------------------------------------------------
// Helpers
// ---------------------------------------------------------------------------
__global__ void f32_to_bf16_kernel(const float* __restrict__ in, bf16* __restrict__ out, int n) {
    int i = (blockIdx.x * blockDim.x + threadIdx.x) * 8;
    if (i < n) {
        float4 a = *(const float4*)(in + i);
        float4 b = *(const float4*)(in + i + 4);
        union { v8bf v; bf16 e[8]; } o;
        o.e[0] = (bf16)a.x; o.e[1] = (bf16)a.y; o.e[2] = (bf16)a.z; o.e[3] = (bf16)a.w;
        o.e[4] = (bf16)b.x; o.e[5] = (bf16)b.y; o.e[6] = (bf16)b.z; o.e[7] = (bf16)b.w;
        *(v8bf*)(out + i) = o.v;
    }
}

// w: (FS, 256, 3) f32  ->  wp: [k][f][d] bf16
__global__ void pack_w_kernel(const float* __restrict__ w, bf16* __restrict__ wp) {
    int i = blockIdx.x * blockDim.x + threadIdx.x;    // 0 .. 196607
    if (i < 3 * FSZ * DD) {
        int k = i >> 16;            // / 65536
        int f = (i >> 8) & 255;
        int d = i & 255;
        wp[i] = (bf16)w[((size_t)f * DD + d) * 3 + k];
    }
}

__global__ void cumsum_kernel(const int* __restrict__ dt, int* __restrict__ csum) {
    int b = threadIdx.x;
    if (b < BB) {
        int s = 0;
        for (int i = 0; i < SS; ++i) { s += dt[b * SS + i]; csum[b * SS + i] = s; }
    }
}

// length_regulate: gather + zero invalid; writes f32 master + bf16 shadow
// block = 256 threads = 4 rows x 64 quad-channels
__global__ void regulate_kernel(const float* __restrict__ x, const int* __restrict__ csum,
                                const unsigned char* __restrict__ melmask,
                                float* __restrict__ xa, bf16* __restrict__ xab) {
    int bt = blockIdx.x * 4 + (threadIdx.x >> 6);   // 0 .. B*T-1
    int c4 = (threadIdx.x & 63) * 4;                // channel base
    int b = bt >> 11;                               // T = 2048
    int t = bt & (TT - 1);
    const int* c = csum + b * SS;
    int lo = 0, hi = SS;                 // searchsorted right: first i with c[i] > t
    while (lo < hi) { int mid = (lo + hi) >> 1; if (c[mid] <= t) lo = mid + 1; else hi = mid; }
    int idx = lo < (SS - 1) ? lo : (SS - 1);
    float4 v = make_float4(0.f, 0.f, 0.f, 0.f);
    if (!melmask[(size_t)b * TT + t])
        v = *(const float4*)(x + ((size_t)b * SS + idx) * DD + c4);
    size_t o = (size_t)bt * DD + c4;
    *(float4*)(xa + o) = v;
    v4bf e; e[0] = (bf16)v.x; e[1] = (bf16)v.y; e[2] = (bf16)v.z; e[3] = (bf16)v.w;
    *(v4bf*)(xab + o) = e;
}

// x_ad += conv1d(target[:, :, None], w(D,1,3), b(D)); 4 channels per thread
__global__ void add_embed_kernel(const float* __restrict__ xa_in, const float* __restrict__ tgt,
                                 const float* __restrict__ w, const float* __restrict__ bias,
                                 float* __restrict__ out_f32, bf16* __restrict__ out_bf16) {
    int bt = blockIdx.x * 4 + (threadIdx.x >> 6);
    int c4 = (threadIdx.x & 63) * 4;
    int b = bt >> 11;
    int t = bt & (TT - 1);
    float tv[3];
    #pragma unroll
    for (int k = 0; k < 3; ++k) {
        int tt = t + k - 1;
        tv[k] = (tt >= 0 && tt < TT) ? tgt[(size_t)b * TT + tt] : 0.f;
    }
    size_t o = (size_t)bt * DD + c4;
    float4 xi = *(const float4*)(xa_in + o);
    float4 bs = *(const float4*)(bias + c4);
    float r[4] = { xi.x + bs.x, xi.y + bs.y, xi.z + bs.z, xi.w + bs.w };
    #pragma unroll
    for (int j = 0; j < 4; ++j) {
        int d = c4 + j;
        #pragma unroll
        for (int k = 0; k < 3; ++k) r[j] += w[d * 3 + k] * tv[k];
    }
    *(float4*)(out_f32 + o) = make_float4(r[0], r[1], r[2], r[3]);
    if (out_bf16) {
        v4bf e; e[0] = (bf16)r[0]; e[1] = (bf16)r[1]; e[2] = (bf16)r[2]; e[3] = (bf16)r[3];
        *(v4bf*)(out_bf16 + o) = e;
    }
}

__global__ void dur_out_kernel(const int* __restrict__ dt, float* __restrict__ out) {
    int i = blockIdx.x * blockDim.x + threadIdx.x;
    if (i < BB * SS) out[i] = (float)dt[i];
}

// ---------------------------------------------------------------------------
// Host launch
// ---------------------------------------------------------------------------
struct Params {
    const float *c1w, *c1b, *ln1g, *ln1b, *c2w, *c2b, *ln2g, *ln2b, *lw, *lb;
};

extern "C" void kernel_launch(void* const* d_in, const int* in_sizes, int n_in,
                              void* d_out, int out_size, void* d_ws, size_t ws_size,
                              hipStream_t stream) {
    (void)in_sizes; (void)n_in; (void)out_size; (void)ws_size;

    const float* x = (const float*)d_in[0];
    auto getP = [&](int base) {
        Params p;
        p.c1w  = (const float*)d_in[base + 0];
        p.c1b  = (const float*)d_in[base + 1];
        p.ln1g = (const float*)d_in[base + 2];
        p.ln1b = (const float*)d_in[base + 3];
        p.c2w  = (const float*)d_in[base + 4];
        p.c2b  = (const float*)d_in[base + 5];
        p.ln2g = (const float*)d_in[base + 6];
        p.ln2b = (const float*)d_in[base + 7];
        p.lw   = (const float*)d_in[base + 8];
        p.lb   = (const float*)d_in[base + 9];
        return p;
    };
    Params dur = getP(1), pit = getP(11), ene = getP(21);
    const float* pe_w = (const float*)d_in[31];
    const float* pe_b = (const float*)d_in[32];
    const float* ee_w = (const float*)d_in[33];
    const float* ee_b = (const float*)d_in[34];
    const float* p_tgt = (const float*)d_in[35];
    const float* e_tgt = (const float*)d_in[36];
    const int*   d_tgt = (const int*)d_in[37];
    const unsigned char* src_mask = (const unsigned char*)d_in[38];
    const unsigned char* mel_mask = (const unsigned char*)d_in[39];

    float* out        = (float*)d_out;
    float* out_xad    = out;                                   // B*T*D
    float* out_ldur   = out_xad + (size_t)BB * TT * DD;        // B*S
    float* out_pitch  = out_ldur + (size_t)BB * SS;            // B*T
    float* out_energy = out_pitch + (size_t)BB * TT;           // B*T
    float* out_dur    = out_energy + (size_t)BB * TT;          // B*S

    // carve workspace
    char* wsp = (char*)d_ws;
    auto alloc = [&](size_t bytes) {
        void* p = (void*)wsp;
        wsp += (bytes + 255) & ~(size_t)255;
        return p;
    };
    bf16* xb  = (bf16*)alloc((size_t)BB * SS * DD * sizeof(bf16));
    bf16* wp[6];
    for (int i = 0; i < 6; ++i) wp[i] = (bf16*)alloc((size_t)3 * FSZ * DD * sizeof(bf16));
    bf16* hb  = (bf16*)alloc((size_t)BB * TT * FSZ * sizeof(bf16));
    float* xa = (float*)alloc((size_t)BB * TT * DD * sizeof(float));
    bf16* xab = (bf16*)alloc((size_t)BB * TT * DD * sizeof(bf16));
    int* csum = (int*)alloc((size_t)BB * SS * sizeof(int));

    // conversions / packing
    {
        int n = BB * SS * DD;
        f32_to_bf16_kernel<<<(n / 8 + 255) / 256, 256, 0, stream>>>(x, xb, n);
    }
    const float* wsrc[6] = { dur.c1w, dur.c2w, pit.c1w, pit.c2w, ene.c1w, ene.c2w };
    for (int i = 0; i < 6; ++i)
        pack_w_kernel<<<(3 * FSZ * DD + 255) / 256, 256, 0, stream>>>(wsrc[i], wp[i]);

    // ---- duration predictor on x (seq = S) ----
    dim3 gS(SS / 32, BB), gT(TT / 32, BB);
    conv_ln_kernel<<<gS, 256, 0, stream>>>(xb, SS, wp[0], dur.c1b, dur.ln1g, dur.ln1b,
                                           hb, nullptr, nullptr, nullptr, nullptr);
    conv_ln_kernel<<<gS, 256, 0, stream>>>(hb, SS, wp[1], dur.c2b, dur.ln2g, dur.ln2b,
                                           nullptr, out_ldur, dur.lw, dur.lb, src_mask);

    // ---- length regulation ----
    cumsum_kernel<<<1, 32, 0, stream>>>(d_tgt, csum);
    regulate_kernel<<<BB * TT / 4, 256, 0, stream>>>(x, csum, mel_mask, xa, xab);

    // ---- pitch predictor on x_ad (seq = T) ----
    conv_ln_kernel<<<gT, 256, 0, stream>>>(xab, TT, wp[2], pit.c1b, pit.ln1g, pit.ln1b,
                                           hb, nullptr, nullptr, nullptr, nullptr);
    conv_ln_kernel<<<gT, 256, 0, stream>>>(hb, TT, wp[3], pit.c2b, pit.ln2g, pit.ln2b,
                                           nullptr, out_pitch, pit.lw, pit.lb, mel_mask);

    // ---- x_ad += pitch embedding ----
    add_embed_kernel<<<BB * TT / 4, 256, 0, stream>>>(xa, p_tgt, pe_w, pe_b, xa, xab);

    // ---- energy predictor ----
    conv_ln_kernel<<<gT, 256, 0, stream>>>(xab, TT, wp[4], ene.c1b, ene.ln1g, ene.ln1b,
                                           hb, nullptr, nullptr, nullptr, nullptr);
    conv_ln_kernel<<<gT, 256, 0, stream>>>(hb, TT, wp[5], ene.c2b, ene.ln2g, ene.ln2b,
                                           nullptr, out_energy, ene.lw, ene.lb, mel_mask);

    // ---- x_ad += energy embedding, final output ----
    add_embed_kernel<<<BB * TT / 4, 256, 0, stream>>>(xa, e_tgt, ee_w, ee_b, out_xad, nullptr);

    // ---- duration passthrough ----
    dur_out_kernel<<<(BB * SS + 255) / 256, 256, 0, stream>>>(d_tgt, out_dur);
}
